// NNConvDecoder_45320494907735
// MI455X (gfx1250) — matrix operand order repaired and checked by reading, not compiled, not run on hardware
//
#include <hip/hip_runtime.h>

typedef __attribute__((ext_vector_type(16))) __bf16 v16bf;
typedef __attribute__((ext_vector_type(8)))  float  v8f;

#define HIDN 64
#define NN 10000
#define NE 20000
#define NG 16
#define KKB 66          // (32*64 + 64) / 32  K-steps for the big edge GEMM

// ---------------- weight pre-swizzle into WMMA B-fragment order (bf16) ----------------
// B layout used here (contiguous k, same as A): lanes 0-15 cols N=0..15 hold K=0..15 (elem j -> K=j),
// lanes 16-31 hold K=16..31.  Each lane's 16 elements stored contiguously:
// out[((kk*4 + nt)*32 + lane)*16 + j]
__global__ void swz_wbig_k(const float* __restrict__ w2, const float* __restrict__ b2,
                           __bf16* __restrict__ out) {
  int idx = blockIdx.x * blockDim.x + threadIdx.x;
  if (idx >= KKB * 4 * 32 * 16) return;
  int j = idx & 15, lane = (idx >> 4) & 31, nt = (idx >> 9) & 3, kk = idx >> 11;
  int kb = (lane < 16) ? 0 : 16;
  int Kg = kk * 32 + kb + j;
  int o  = nt * 16 + (lane & 15);
  // rows 0..2047 of Wbig are w2 flat ( (k*64+i)*64+o == k*4096+i*64+o ), rows 2048.. are b2
  float v = (Kg < 2048) ? w2[Kg * 64 + o] : b2[(Kg - 2048) * 64 + o];
  out[idx] = (__bf16)v;
}

__global__ void swz_mat64_k(const float* __restrict__ m, __bf16* __restrict__ out) {
  int idx = blockIdx.x * blockDim.x + threadIdx.x;
  if (idx >= 2 * 4 * 32 * 16) return;
  int j = idx & 15, lane = (idx >> 4) & 31, nt = (idx >> 9) & 3, kk = idx >> 11;
  int kb = (lane < 16) ? 0 : 16;
  int Kg = kk * 32 + kb + j;
  int o  = nt * 16 + (lane & 15);
  out[idx] = (__bf16)m[Kg * 64 + o];
}

// ---------------- small helpers ----------------
__global__ void zero_f32_k(float* __restrict__ p, int n) {
  for (int i = blockIdx.x * blockDim.x + threadIdx.x; i < n; i += gridDim.x * blockDim.x)
    p[i] = 0.f;
}

__global__ void lift_k(const float* __restrict__ x, const float* __restrict__ w,
                       const float* __restrict__ b, float* __restrict__ h) {
  int idx = blockIdx.x * blockDim.x + threadIdx.x;
  if (idx >= NN * HIDN) return;
  int n = idx >> 6, o = idx & 63;
  float acc = b[o];
#pragma unroll
  for (int i = 0; i < 5; ++i) acc += x[n * 5 + i] * w[i * 64 + o];
  h[idx] = fmaxf(acc, 0.f);
}

__global__ void edge_mlp_k(const float* __restrict__ ea, const float* __restrict__ w1,
                           const float* __restrict__ b1, float* __restrict__ he) {
  int idx = blockIdx.x * blockDim.x + threadIdx.x;
  if (idx >= NE * 32) return;
  int e = idx >> 5, k = idx & 31;
  he[idx] = fmaxf(ea[e] * w1[k] + b1[k], 0.f);
}

__global__ void count_deg_k(const int* __restrict__ dst, float* __restrict__ deg) {
  int e = blockIdx.x * blockDim.x + threadIdx.x;
  if (e < NE) atomicAdd(&deg[dst[e]], 1.f);
}

__global__ void count_batch_k(const int* __restrict__ batch, float* __restrict__ gcnt) {
  int n = blockIdx.x * blockDim.x + threadIdx.x;
  if (n < NN) atomicAdd(&gcnt[batch[n]], 1.f);
}

// ---------------- fused edge-message GEMM  msg = u @ Wbig , scattered to agg[dst] ------
// block = 128 threads = 4 waves; 64 edges/block; each wave owns a 16-edge M-tile and
// iterates the 4 N-tiles. K = 2112 = 66 bf16 WMMA steps. Per lane only TWO distinct
// 16-elem x-fragments exist across all K-steps -> hoist them to registers as bf16, so
// the inner loop is: 1 scalar LDS read (h_e) per 8 WMMAs + 8 v_pk_mul_bf16 per WMMA pair.
__global__ __launch_bounds__(128) void msg_gemm_k(
    const float* __restrict__ hin, const float* __restrict__ he,
    const __bf16* __restrict__ wswz, const int* __restrict__ src,
    const int* __restrict__ dst, float* __restrict__ agg) {
  __shared__ float x_s[64][64];   // gathered source-node features
  __shared__ float he_s[64][32];  // edge-MLP hidden
  __shared__ int   dst_s[64];

  int tid = threadIdx.x;
  int e0  = blockIdx.x * 64;
  for (int t = tid; t < 64 * 16; t += 128) {       // float4-vectorized gather
    int r = t >> 4, q = t & 15, e = e0 + r;
    float4 v = make_float4(0.f, 0.f, 0.f, 0.f);
    if (e < NE) v = ((const float4*)(hin + src[e] * 64))[q];
    ((float4*)&x_s[r][0])[q] = v;
  }
  for (int t = tid; t < 64 * 8; t += 128) {
    int r = t >> 3, q = t & 7, e = e0 + r;
    float4 v = make_float4(0.f, 0.f, 0.f, 0.f);
    if (e < NE) v = ((const float4*)(he + e * 32))[q];
    ((float4*)&he_s[r][0])[q] = v;
  }
  if (tid < 64) dst_s[tid] = (e0 + tid < NE) ? dst[e0 + tid] : 0;
  __syncthreads();

  int wave = tid >> 5;
  int lane = tid & 31;
  int row  = (wave << 4) + (lane & 15);      // A-matrix M for this lane
  int kbA  = (lane < 16) ? 0 : 16;           // K sub-base per lane half

  // hoist the two per-lane x fragments (bf16) once
  v16bf xf[2];
#pragma unroll
  for (int p = 0; p < 2; ++p)
#pragma unroll
    for (int j = 0; j < 16; ++j) {
      int i = (p << 5) + kbA + j;
      xf[p][j] = (__bf16)x_s[row][i];
    }

  const __bf16* bp = wswz + (lane << 4);
  v8f c[4] = {};

  for (int kp = 0; kp < 32; ++kp) {          // kp = kk/2 ; both parities share h_e[kp]
    __bf16 hvb = (__bf16)he_s[row][kp];
    v16bf hs;
#pragma unroll
    for (int j = 0; j < 16; ++j) hs[j] = hvb;
#pragma unroll
    for (int p = 0; p < 2; ++p) {
      v16bf a = hs * xf[p];                  // v_pk_mul_bf16 x8
#pragma unroll
      for (int nt = 0; nt < 4; ++nt) {
        v16bf b = *(const v16bf*)(bp + (size_t)((kp * 2 + p) * 4 + nt) * 512);
        c[nt] = __builtin_amdgcn_wmma_f32_16x16x32_bf16(false, a, false, b, (short)0,
                                                        c[nt], false, false);
      }
    }
  }
  // K-steps 64..65: bias rows of Wbig, A is the raw x fragment
#pragma unroll
  for (int p = 0; p < 2; ++p) {
#pragma unroll
    for (int nt = 0; nt < 4; ++nt) {
      v16bf b = *(const v16bf*)(bp + (size_t)((64 + p) * 4 + nt) * 512);
      c[nt] = __builtin_amdgcn_wmma_f32_16x16x32_bf16(false, xf[p], false, b, (short)0,
                                                      c[nt], false, false);
    }
  }

  // scatter-add messages to destination nodes (C/D layout: elem r -> M = r + 8*(lane>=16))
  int col   = lane & 15;
  int rbase = (lane < 16) ? 0 : 8;
#pragma unroll
  for (int nt = 0; nt < 4; ++nt) {
#pragma unroll
    for (int r = 0; r < 8; ++r) {
      int lrow = (wave << 4) + rbase + r;
      int e    = e0 + lrow;
      if (e < NE) atomicAdd(&agg[dst_s[lrow] * 64 + nt * 16 + col], c[nt][r]);
    }
  }
}

// ---------------- node update: relu( h@root_w + agg/deg + bias ) via WMMA ----------------
__global__ __launch_bounds__(128) void node_upd_k(
    const float* __restrict__ hin, const float* __restrict__ agg,
    const __bf16* __restrict__ rswz, const float* __restrict__ bias,
    const float* __restrict__ deg, float* __restrict__ hout) {
  __shared__ float h_s[64][64];
  __shared__ float rd_s[64];
  int tid = threadIdx.x;
  int n0  = blockIdx.x * 64;
  for (int t = tid; t < 64 * 16; t += 128) {
    int r = t >> 4, q = t & 15, n = n0 + r;
    float4 v = make_float4(0.f, 0.f, 0.f, 0.f);
    if (n < NN) v = ((const float4*)(hin + n * 64))[q];
    ((float4*)&h_s[r][0])[q] = v;
  }
  if (tid < 64) {
    int n = n0 + tid;
    rd_s[tid] = (n < NN) ? 1.f / fmaxf(deg[n], 1.f) : 0.f;
  }
  __syncthreads();

  int wave = tid >> 5, lane = tid & 31;
  int row = (wave << 4) + (lane & 15);
  int kbA = (lane < 16) ? 0 : 16;

  v16bf af[2];
#pragma unroll
  for (int p = 0; p < 2; ++p)
#pragma unroll
    for (int j = 0; j < 16; ++j) {
      int K = (p << 5) + kbA + j;
      af[p][j] = (__bf16)h_s[row][K];
    }

  const __bf16* bp = rswz + (lane << 4);
  v8f c[4] = {};
#pragma unroll
  for (int kk = 0; kk < 2; ++kk) {
#pragma unroll
    for (int nt = 0; nt < 4; ++nt) {
      v16bf b = *(const v16bf*)(bp + (size_t)(kk * 4 + nt) * 512);
      c[nt] = __builtin_amdgcn_wmma_f32_16x16x32_bf16(false, af[kk], false, b, (short)0,
                                                      c[nt], false, false);
    }
  }

  int col = lane & 15;
  int rbase = (lane < 16) ? 0 : 8;
#pragma unroll
  for (int nt = 0; nt < 4; ++nt) {
#pragma unroll
    for (int r = 0; r < 8; ++r) {
      int lrow = (wave << 4) + rbase + r;
      int n = n0 + lrow;
      if (n < NN) {
        int o = nt * 16 + col;
        float v = c[nt][r] + agg[n * 64 + o] * rd_s[lrow] + bias[o];
        hout[n * 64 + o] = fmaxf(v, 0.f);
      }
    }
  }
}

// ---------------- pooling + head ----------------
__global__ void pool_k(const float* __restrict__ h, const int* __restrict__ batch,
                       float* __restrict__ gsum) {
  int idx = blockIdx.x * blockDim.x + threadIdx.x;
  if (idx >= NN * 64) return;
  int n = idx >> 6, o = idx & 63;
  atomicAdd(&gsum[batch[n] * 64 + o], h[idx]);
}

__global__ __launch_bounds__(128) void head_k(
    const float* __restrict__ gsum, const float* __restrict__ gcnt,
    const float* __restrict__ w1, const float* __restrict__ b1,
    const float* __restrict__ w2, const float* __restrict__ b2, float* __restrict__ out) {
  __shared__ float hid[NG * 64];
  int tid = threadIdx.x;
  for (int t = tid; t < NG * 64; t += 128) {
    int gr = t >> 6, hcol = t & 63;
    float cm = 1.f / fmaxf(gcnt[gr], 1.f);
    float acc = b1[hcol];
    for (int i = 0; i < 64; ++i) {
      float s = gsum[gr * 64 + i];
      acc += s * cm * w1[i * 64 + hcol] + s * w1[(64 + i) * 64 + hcol];
    }
    hid[t] = fmaxf(acc, 0.f);
  }
  __syncthreads();
  if (tid < NG) {
    float acc = b2[0];
    for (int t = 0; t < 64; ++t) acc += hid[tid * 64 + t] * w2[t];
    out[tid] = acc;
  }
}

// ---------------- launch ----------------
extern "C" void kernel_launch(void* const* d_in, const int* in_sizes, int n_in,
                              void* d_out, int out_size, void* d_ws, size_t ws_size,
                              hipStream_t stream) {
  const float* x       = (const float*)d_in[0];
  const int*   eidx    = (const int*)d_in[1];
  const float* eattr   = (const float*)d_in[2];
  const int*   batch   = (const int*)d_in[3];
  const float* lift_w  = (const float*)d_in[4];
  const float* lift_b  = (const float*)d_in[5];
  const float* head_w1 = (const float*)d_in[6];
  const float* head_b1 = (const float*)d_in[7];
  const float* head_w2 = (const float*)d_in[8];
  const float* head_b2 = (const float*)d_in[9];
  const float* mw1[2] = {(const float*)d_in[10], (const float*)d_in[16]};
  const float* mb1[2] = {(const float*)d_in[11], (const float*)d_in[17]};
  const float* mw2[2] = {(const float*)d_in[12], (const float*)d_in[18]};
  const float* mb2[2] = {(const float*)d_in[13], (const float*)d_in[19]};
  const float* rtw[2] = {(const float*)d_in[14], (const float*)d_in[20]};
  const float* lbs[2] = {(const float*)d_in[15], (const float*)d_in[21]};
  const int* src = eidx;
  const int* dst = eidx + NE;

  // workspace layout (floats), everything 64B aligned; total ~10.9 MB
  float* hA   = (float*)d_ws;          // [NN*64]
  float* hB   = hA + 640000;           // [NN*64]
  float* he   = hB + 640000;           // [NE*32]
  float* agg  = he + 640000;           // [NN*64]
  float* deg  = agg + 640000;          // [10016]
  float* gsum = deg + 10016;           // [1024]
  float* gcnt = gsum + 1024;           // [64]
  __bf16* wswz[2];
  wswz[0] = (__bf16*)(gcnt + 64);      // [66*4*32*16]
  wswz[1] = wswz[0] + KKB * 2048;
  __bf16* rswz[2];
  rswz[0] = wswz[1] + KKB * 2048;      // [2*4*32*16]
  rswz[1] = rswz[0] + 4096;

  // 1) weight pre-swizzle (bf16)
  for (int l = 0; l < 2; ++l) {
    swz_wbig_k<<<(KKB * 2048 + 255) / 256, 256, 0, stream>>>(mw2[l], mb2[l], wswz[l]);
    swz_mat64_k<<<16, 256, 0, stream>>>(rtw[l], rswz[l]);
  }
  // 2) zero deg/gsum/gcnt (contiguous) ; 3) lift ; 4) counts
  zero_f32_k<<<44, 256, 0, stream>>>(deg, 10016 + 1024 + 64);
  lift_k<<<(NN * 64 + 255) / 256, 256, 0, stream>>>(x, lift_w, lift_b, hA);
  count_deg_k<<<(NE + 255) / 256, 256, 0, stream>>>(dst, deg);
  count_batch_k<<<(NN + 255) / 256, 256, 0, stream>>>(batch, gcnt);

  // 5) two NNConv layers: hA -> hB -> hA
  float* hio[3] = {hA, hB, hA};
  for (int l = 0; l < 2; ++l) {
    edge_mlp_k<<<(NE * 32 + 255) / 256, 256, 0, stream>>>(eattr, mw1[l], mb1[l], he);
    zero_f32_k<<<(NN * 64 + 255) / 256, 256, 0, stream>>>(agg, NN * 64);
    msg_gemm_k<<<(NE + 63) / 64, 128, 0, stream>>>(hio[l], he, wswz[l], src, dst, agg);
    node_upd_k<<<(NN + 63) / 64, 128, 0, stream>>>(hio[l], agg, rswz[l], lbs[l], deg,
                                                   hio[l + 1]);
  }

  // 6) pool + head
  pool_k<<<(NN * 64 + 255) / 256, 256, 0, stream>>>(hA, batch, gsum);
  head_k<<<1, 128, 0, stream>>>(gsum, gcnt, head_w1, head_b1, head_w2, head_b2,
                                (float*)d_out);
}
